// AttentionBlock_61753039782643
// MI455X (gfx1250) — matrix operand order repaired and checked by reading, not compiled, not run on hardware
//
#include <hip/hip_runtime.h>

typedef __attribute__((ext_vector_type(16))) _Float16 v16h;
typedef __attribute__((ext_vector_type(8)))  _Float16 v8h;
typedef __attribute__((ext_vector_type(8)))  float    v8f;

#define NB 4
#define HW 1024
#define CH 512
#define NHEAD 8
#define HD 64

static __device__ __forceinline__ v16h cat8(v8h lo, v8h hi) {
  return __builtin_shufflevector(lo, hi, 0,1,2,3,4,5,6,7,8,9,10,11,12,13,14,15);
}

static __device__ __forceinline__ v8f wmma_f16(v16h a, v16h b, v8f c) {
  // 8 args: (neg_a, A, neg_b, B, c_mod, C, reuse_a, reuse_b)
  return __builtin_amdgcn_wmma_f32_16x16x32_f16(false, a, false, b, (short)0, c, false, false);
}

// ---------------- 1. GroupNorm stats: one block per (b, group) ----------------
__global__ void gn_stats_kernel(const float* __restrict__ x, float* __restrict__ stats) {
  int bg = blockIdx.x;                 // b*32 + g
  int b = bg >> 5, g = bg & 31;
  int tid = threadIdx.x;
  int c = tid & 15, hw0 = tid >> 4;
  const float* base = x + ((size_t)b * HW) * CH + g * 16 + c;
  float s = 0.f, s2 = 0.f;
  for (int hw = hw0; hw < HW; hw += 16) {
    float v = base[(size_t)hw * CH];
    s += v; s2 += v * v;
  }
  __shared__ float ssum[256], ssq[256];
  ssum[tid] = s; ssq[tid] = s2;
  __syncthreads();
  for (int off = 128; off > 0; off >>= 1) {
    if (tid < off) { ssum[tid] += ssum[tid + off]; ssq[tid] += ssq[tid + off]; }
    __syncthreads();
  }
  if (tid == 0) {
    float mean = ssum[0] * (1.0f / 16384.0f);
    float var  = ssq[0]  * (1.0f / 16384.0f) - mean * mean;
    stats[bg * 2]     = mean;
    stats[bg * 2 + 1] = rsqrtf(var + 1e-6f);
  }
}

// ---------------- 2. Normalize + affine -> f16 ----------------
__global__ void gn_norm_kernel(const float* __restrict__ x, const float* __restrict__ stats,
                               const float* __restrict__ gsc, const float* __restrict__ gbi,
                               _Float16* __restrict__ xn) {
  size_t idx = (size_t)blockIdx.x * blockDim.x + threadIdx.x;   // < 4096*512
  int m  = (int)(idx >> 9);
  int ch = (int)(idx & 511);
  int b = m >> 10, g = ch >> 4;
  float mean = stats[(b * 32 + g) * 2];
  float rstd = stats[(b * 32 + g) * 2 + 1];
  xn[idx] = (_Float16)((x[idx] - mean) * rstd * gsc[ch] + gbi[ch]);
}

// ---------------- 3. f32 weight -> f16 transposed (B^T row-major) ----------------
__global__ void cvt_transpose_kernel(const float* __restrict__ src, _Float16* __restrict__ dst,
                                     int rows, int cols) {
  int idx = blockIdx.x * blockDim.x + threadIdx.x;
  if (idx >= rows * cols) return;
  int r = idx / cols, c = idx - r * cols;
  dst[(size_t)c * rows + r] = (_Float16)src[idx];
}

// ---------------- 4. QKV GEMM: [4096,512] x [512,1536], scatter q/k/vT ----------------
__global__ void __launch_bounds__(256) qkv_gemm_kernel(
    const _Float16* __restrict__ xn, const _Float16* __restrict__ wT,
    _Float16* __restrict__ qo, _Float16* __restrict__ ko, _Float16* __restrict__ vT) {
  int wave = threadIdx.x >> 5, lane = threadIdx.x & 31;
  int half = lane >> 4, lq = lane & 15;
  int mBase  = blockIdx.x * 128 + wave * 16;
  int nStrip = blockIdx.y * 64;
  v8f acc[4] = {};
  const _Float16* arow = xn + (size_t)(mBase + lq) * CH;
  for (int kk = 0; kk < CH; kk += 32) {
    v16h a = cat8(*(const v8h*)(arow + kk + half * 16),
                  *(const v8h*)(arow + kk + half * 16 + 8));
#pragma unroll
    for (int t = 0; t < 4; ++t) {
      const _Float16* bp = wT + (size_t)(nStrip + t * 16 + lq) * CH + kk + half * 16;
      acc[t] = wmma_f16(a, *(const v16h*)bp, acc[t]);
    }
  }
  const float scale = 0.04419417382415922f;   // C^-0.5, folded into q
#pragma unroll
  for (int t = 0; t < 4; ++t) {
    int col = nStrip + t * 16 + lq;
    int h = col / 192;
    int rem = col - h * 192;
    int typ = rem >> 6;       // 0=q 1=k 2=v (16-wide tile never straddles)
    int e = rem & 63;
#pragma unroll
    for (int r = 0; r < 8; ++r) {
      int m = mBase + r + half * 8;
      float val = acc[t][r];
      if (typ == 0) {
        qo[(size_t)m * CH + h * HD + e] = (_Float16)(val * scale);
      } else if (typ == 1) {
        ko[(size_t)m * CH + h * HD + e] = (_Float16)val;
      } else {
        int b = m >> 10, nloc = m & 1023;
        vT[((size_t)(b * NHEAD + h) * HD + e) * HW + nloc] = (_Float16)val;
      }
    }
  }
}

// ---------------- 5. Flash attention: one wave = 16 query rows of one (b,h) ----------------
__global__ void __launch_bounds__(256) attn_kernel(
    const _Float16* __restrict__ q, const _Float16* __restrict__ k,
    const _Float16* __restrict__ vT, _Float16* __restrict__ o) {
  int bh = blockIdx.x;
  int b = bh >> 3, h = bh & 7;
  int wave = threadIdx.x >> 5, lane = threadIdx.x & 31;
  int half = lane >> 4, lq = lane & 15;
  int iBase = blockIdx.y * 128 + wave * 16;

  __shared__ __align__(16) _Float16 ldsP[8][16 * 32];   // per-wave 1KB P tile
  _Float16* myP = ldsP[wave];

  const _Float16* qrow = q + (size_t)(b * HW + iBase + lq) * CH + h * HD;
  v16h a0 = cat8(*(const v8h*)(qrow + half * 16),      *(const v8h*)(qrow + 8 + half * 16));
  v16h a1 = cat8(*(const v8h*)(qrow + 32 + half * 16), *(const v8h*)(qrow + 40 + half * 16));

  float mrow[8], lrow[8];
#pragma unroll
  for (int r = 0; r < 8; ++r) { mrow[r] = -1e30f; lrow[r] = 0.f; }
  v8f acc[4] = {};

  for (int j0 = 0; j0 < HW; j0 += 32) {
    v8f s0 = {}, s1 = {};
    {
      const _Float16* kr0 = k + (size_t)(b * HW + j0 + lq) * CH + h * HD + half * 16;
      s0 = wmma_f16(a0, *(const v16h*)(kr0), s0);
      s0 = wmma_f16(a1, *(const v16h*)(kr0 + 32), s0);
      const _Float16* kr1 = kr0 + 16 * CH;
      s1 = wmma_f16(a0, *(const v16h*)(kr1), s1);
      s1 = wmma_f16(a1, *(const v16h*)(kr1 + 32), s1);
    }
    float rmax[8];
#pragma unroll
    for (int r = 0; r < 8; ++r) rmax[r] = fmaxf(s0[r], s1[r]);
#pragma unroll
    for (int off = 8; off > 0; off >>= 1)
#pragma unroll
      for (int r = 0; r < 8; ++r) rmax[r] = fmaxf(rmax[r], __shfl_xor(rmax[r], off));

    float p0[8], p1[8], rsum[8];
#pragma unroll
    for (int r = 0; r < 8; ++r) {
      float mn   = fmaxf(mrow[r], rmax[r]);
      float corr = __expf(mrow[r] - mn);
      mrow[r] = mn;
      p0[r] = __expf(s0[r] - mn);
      p1[r] = __expf(s1[r] - mn);
      rsum[r] = p0[r] + p1[r];
      lrow[r] *= corr;
#pragma unroll
      for (int n = 0; n < 4; ++n) acc[n][r] *= corr;
    }
#pragma unroll
    for (int off = 8; off > 0; off >>= 1)
#pragma unroll
      for (int r = 0; r < 8; ++r) rsum[r] += __shfl_xor(rsum[r], off);
#pragma unroll
    for (int r = 0; r < 8; ++r) lrow[r] += rsum[r];

    // C/D layout -> LDS (row major 16x32 f16) -> A layout (same-wave DS ordering)
#pragma unroll
    for (int r = 0; r < 8; ++r) {
      int row = r + half * 8;
      myP[row * 32 + lq]      = (_Float16)p0[r];
      myP[row * 32 + 16 + lq] = (_Float16)p1[r];
    }
    v16h pa = cat8(*(const v8h*)(myP + lq * 32 + half * 16),
                   *(const v8h*)(myP + lq * 32 + half * 16 + 8));
#pragma unroll
    for (int n = 0; n < 4; ++n) {
      const _Float16* vr = vT + ((size_t)bh * HD + n * 16 + lq) * HW + j0 + half * 16;
      acc[n] = wmma_f16(pa, *(const v16h*)vr, acc[n]);
    }
  }
#pragma unroll
  for (int n = 0; n < 4; ++n)
#pragma unroll
    for (int r = 0; r < 8; ++r) {
      int row = iBase + r + half * 8;
      o[(size_t)(b * HW + row) * CH + h * HD + n * 16 + lq] = (_Float16)(acc[n][r] / lrow[r]);
    }
}

// ---------------- 6. Output projection + bias + residual ----------------
__global__ void __launch_bounds__(256) out_gemm_kernel(
    const _Float16* __restrict__ ain, const _Float16* __restrict__ wT,
    const float* __restrict__ bias, const float* __restrict__ x,
    float* __restrict__ out) {
  int wave = threadIdx.x >> 5, lane = threadIdx.x & 31;
  int half = lane >> 4, lq = lane & 15;
  int mBase  = blockIdx.x * 128 + wave * 16;
  int nStrip = blockIdx.y * 64;
  v8f acc[4] = {};
  const _Float16* arow = ain + (size_t)(mBase + lq) * CH;
  for (int kk = 0; kk < CH; kk += 32) {
    v16h a = cat8(*(const v8h*)(arow + kk + half * 16),
                  *(const v8h*)(arow + kk + half * 16 + 8));
#pragma unroll
    for (int t = 0; t < 4; ++t) {
      const _Float16* bp = wT + (size_t)(nStrip + t * 16 + lq) * CH + kk + half * 16;
      acc[t] = wmma_f16(a, *(const v16h*)bp, acc[t]);
    }
  }
#pragma unroll
  for (int t = 0; t < 4; ++t) {
    int col = nStrip + t * 16 + lq;
    float bv = bias[col];
#pragma unroll
    for (int r = 0; r < 8; ++r) {
      int m = mBase + r + half * 8;
      out[(size_t)m * CH + col] = acc[t][r] + bv + x[(size_t)m * CH + col];
    }
  }
}

// ---------------- workspace layout (bytes, all 32B-aligned) ----------------
static constexpr size_t O_STATS = 0;                             // 1 KB
static constexpr size_t O_XN    = 4096;                          // 4 MB  f16 [4096,512]
static constexpr size_t O_WQKVT = O_XN    + (size_t)4096*512*2;  // 1.5MB f16 [1536,512]
static constexpr size_t O_WOUTT = O_WQKVT + (size_t)1536*512*2;  // 0.5MB f16 [512,512]
static constexpr size_t O_Q     = O_WOUTT + (size_t)512*512*2;   // 4 MB
static constexpr size_t O_K     = O_Q     + (size_t)4096*512*2;  // 4 MB
static constexpr size_t O_VT    = O_K     + (size_t)4096*512*2;  // 4 MB  f16 [32,64,1024]
static constexpr size_t O_ATTN  = O_VT    + (size_t)4096*512*2;  // 4 MB

extern "C" void kernel_launch(void* const* d_in, const int* in_sizes, int n_in,
                              void* d_out, int out_size, void* d_ws, size_t ws_size,
                              hipStream_t stream) {
  (void)in_sizes; (void)n_in; (void)out_size; (void)ws_size;
  const float* x    = (const float*)d_in[0];
  const float* gsc  = (const float*)d_in[1];
  const float* gbi  = (const float*)d_in[2];
  const float* wqkv = (const float*)d_in[3];
  const float* wout = (const float*)d_in[4];
  const float* bout = (const float*)d_in[5];
  float* out = (float*)d_out;
  char* ws = (char*)d_ws;

  float*    stats = (float*)   (ws + O_STATS);
  _Float16* xn    = (_Float16*)(ws + O_XN);
  _Float16* wqkvT = (_Float16*)(ws + O_WQKVT);
  _Float16* woutT = (_Float16*)(ws + O_WOUTT);
  _Float16* qbuf  = (_Float16*)(ws + O_Q);
  _Float16* kbuf  = (_Float16*)(ws + O_K);
  _Float16* vT    = (_Float16*)(ws + O_VT);
  _Float16* attn  = (_Float16*)(ws + O_ATTN);

  gn_stats_kernel<<<dim3(NB * 32), dim3(256), 0, stream>>>(x, stats);
  gn_norm_kernel<<<dim3((NB * HW * CH) / 256), dim3(256), 0, stream>>>(x, stats, gsc, gbi, xn);
  cvt_transpose_kernel<<<dim3((512 * 1536 + 255) / 256), dim3(256), 0, stream>>>(wqkv, wqkvT, 512, 1536);
  cvt_transpose_kernel<<<dim3((512 * 512 + 255) / 256), dim3(256), 0, stream>>>(wout, woutT, 512, 512);
  qkv_gemm_kernel<<<dim3(32, 24), dim3(256), 0, stream>>>(xn, wqkvT, qbuf, kbuf, vT);
  attn_kernel<<<dim3(32, 8), dim3(256), 0, stream>>>(qbuf, kbuf, vT, attn);
  out_gemm_kernel<<<dim3(32, 8), dim3(256), 0, stream>>>(attn, woutT, bout, x, out);
}
